// Block6_90572270338515
// MI455X (gfx1250) — matrix operand order repaired and checked by reading, not compiled, not run on hardware
//
#include <hip/hip_runtime.h>
#include <hip/hip_bf16.h>
#include <stdint.h>

#define B_   2
#define T_   1024
#define C_   2048
#define H_   32
#define K_   64
#define DMIX 32
#define DDEC 64
#define FF_  7168
#define NTOK (B_ * T_)          // 2048 rows
#define GN_EPS_ 6.4e-4f         // 1e-5 * 8^2

typedef __attribute__((ext_vector_type(16))) __bf16 v16bf;
typedef __attribute__((ext_vector_type(8)))  float  v8f;

// ---------------- helpers ----------------
__device__ __forceinline__ unsigned short f2bf_bits(float f) {
  union { float f; unsigned int u; } c; c.f = f;
  unsigned int u = c.u;
  unsigned int r = (u + 0x7FFFu + ((u >> 16) & 1u)) >> 16;  // RNE
  return (unsigned short)r;
}

union Frag16 { v16bf v; uint4 q[2]; unsigned short h[16]; };

// ---------------- fp32 -> bf16 convert ----------------
__global__ void f2bf_kernel(const float* __restrict__ src,
                            unsigned short* __restrict__ dst, size_t n) {
  size_t i = (size_t)blockIdx.x * blockDim.x + threadIdx.x;
  size_t st = (size_t)gridDim.x * blockDim.x;
  for (; i < n; i += st) dst[i] = f2bf_bits(src[i]);
}

// ---------------- LayerNorm (per token row), optional last-token capture ----
__global__ void ln_kernel(const float* __restrict__ x, const float* __restrict__ w,
                          const float* __restrict__ b, float* __restrict__ y,
                          float* __restrict__ last_out) {
  int row = blockIdx.x;                    // b*T + t
  const float* xr = x + (size_t)row * C_;
  float* yr = y + (size_t)row * C_;
  __shared__ float red[256];
  float s = 0.f, s2 = 0.f;
  for (int c = threadIdx.x; c < C_; c += 256) { float v = xr[c]; s += v; s2 += v * v; }
  red[threadIdx.x] = s; __syncthreads();
  for (int off = 128; off > 0; off >>= 1) {
    if (threadIdx.x < off) red[threadIdx.x] += red[threadIdx.x + off];
    __syncthreads();
  }
  float mean = red[0] / C_; __syncthreads();
  red[threadIdx.x] = s2; __syncthreads();
  for (int off = 128; off > 0; off >>= 1) {
    if (threadIdx.x < off) red[threadIdx.x] += red[threadIdx.x + off];
    __syncthreads();
  }
  float var = red[0] / C_ - mean * mean;
  float inv = rsqrtf(var + 1e-5f);
  int t = row % T_, bb = row / T_;
  for (int c = threadIdx.x; c < C_; c += 256) {
    float v = (xr[c] - mean) * inv * w[c] + b[c];
    yr[c] = v;
    if (last_out && t == T_ - 1) last_out[(size_t)bb * C_ + c] = v;
  }
}

// ---------------- hh = tanh((xn + xx*maa_x) @ maa_w1), 160 outputs/token ----
__global__ void mix_hh_kernel(const float* __restrict__ xn, const float* __restrict__ tshift,
                              const float* __restrict__ maa_x, const float* __restrict__ maa_w1,
                              float* __restrict__ hh) {
  int row = blockIdx.x, t = row % T_, bb = row / T_;
  __shared__ float mixin[C_];
  const float* xr = xn + (size_t)row * C_;
  const float* pv = (t == 0) ? (tshift + (size_t)bb * C_) : (xr - C_);
  for (int c = threadIdx.x; c < C_; c += blockDim.x) {
    float a = xr[c];
    mixin[c] = a + (pv[c] - a) * maa_x[c];
  }
  __syncthreads();
  int j = threadIdx.x;                     // blockDim = 160
  if (j < 5 * DMIX) {
    float acc = 0.f;
    for (int c = 0; c < C_; ++c) acc += mixin[c] * maa_w1[c * (5 * DMIX) + j];
    hh[(size_t)row * (5 * DMIX) + j] = tanhf(acc);
  }
}

// ---------------- apply mm, produce xw(f32) and xk/xv/xr/xg (bf16) ----------
__global__ void mix_apply_kernel(const float* __restrict__ xn, const float* __restrict__ tshift,
                                 const float* __restrict__ hh, const float* __restrict__ maa_w2,
                                 const float* __restrict__ maa_w, const float* __restrict__ maa_k,
                                 const float* __restrict__ maa_v, const float* __restrict__ maa_r,
                                 const float* __restrict__ maa_g,
                                 float* __restrict__ xw, unsigned short* __restrict__ xk,
                                 unsigned short* __restrict__ xv, unsigned short* __restrict__ xr_o,
                                 unsigned short* __restrict__ xg) {
  int row = blockIdx.x, t = row % T_, bb = row / T_;
  __shared__ float hs[5 * DMIX];
  if (threadIdx.x < 5 * DMIX) hs[threadIdx.x] = hh[(size_t)row * (5 * DMIX) + threadIdx.x];
  __syncthreads();
  const float* xrow = xn + (size_t)row * C_;
  const float* pv = (t == 0) ? (tshift + (size_t)bb * C_) : (xrow - C_);
  for (int c = threadIdx.x; c < C_; c += blockDim.x) {
    float a = xrow[c], xx = pv[c] - a;
    float mm[5];
#pragma unroll
    for (int f = 0; f < 5; ++f) {
      float acc = 0.f;
      const float* w2 = maa_w2 + ((size_t)f * DMIX) * C_ + c;
#pragma unroll
      for (int d = 0; d < DMIX; ++d) acc += hs[f * DMIX + d] * w2[(size_t)d * C_];
      mm[f] = acc;
    }
    size_t o = (size_t)row * C_ + c;
    xw[o]   = a + xx * (maa_w[c] + mm[0]);
    xk[o]   = f2bf_bits(a + xx * (maa_k[c] + mm[1]));
    xv[o]   = f2bf_bits(a + xx * (maa_v[c] + mm[2]));
    xr_o[o] = f2bf_bits(a + xx * (maa_r[c] + mm[3]));
    xg[o]   = f2bf_bits(a + xx * (maa_g[c] + mm[4]));
  }
}

// ---------------- tdh = tanh(xw @ td_w1)  (C->64 per token) ----------------
__global__ void td1_kernel(const float* __restrict__ xw, const float* __restrict__ td_w1,
                           float* __restrict__ tdh) {
  int row = blockIdx.x;
  __shared__ float xs[C_];
  for (int c = threadIdx.x; c < C_; c += blockDim.x) xs[c] = xw[(size_t)row * C_ + c];
  __syncthreads();
  int j = threadIdx.x;                     // blockDim = 64
  float acc = 0.f;
  for (int c = 0; c < C_; ++c) acc += xs[c] * td_w1[c * DDEC + j];
  tdh[(size_t)row * DDEC + j] = tanhf(acc);
}

// ---------------- ew = exp(-exp(td + tdh @ td_w2)) -------------------------
__global__ void wdecay_kernel(const float* __restrict__ tdh, const float* __restrict__ td_w2,
                              const float* __restrict__ td, float* __restrict__ ew) {
  int row = blockIdx.x;
  __shared__ float hs[DDEC];
  if (threadIdx.x < DDEC) hs[threadIdx.x] = tdh[(size_t)row * DDEC + threadIdx.x];
  __syncthreads();
  for (int c = threadIdx.x; c < C_; c += blockDim.x) {
    float acc = 0.f;
#pragma unroll
    for (int j = 0; j < DDEC; ++j) acc += hs[j] * td_w2[j * C_ + c];
    ew[(size_t)row * C_ + c] = expf(-expf(td[c] + acc));
  }
}

// ---------------- bf16 WMMA GEMM: Y[M,N] = A[M,Kd] * W[N,Kd]^T --------------
// 256 threads = 8 waves (4 along M x 2 along N); block tile 128x64.
// Each wave holds a 2x2 register tile (32x32 output): per 32-K step it loads
// 4 fragments (2 A strips + 2 B strips, 128B/lane) and issues 4
// v_wmma_f32_16x16x32_bf16 -> 16 flops/byte from L2 (1.5x fewer loads than a
// 1x2 tile). Accumulate fp32.
__global__ void __launch_bounds__(256)
gemm_bf16_wmma(const unsigned short* __restrict__ A, const unsigned short* __restrict__ W,
               float* __restrict__ Y, int M, int N, int Kd) {
  const int lane = threadIdx.x & 31;
  const int wave = threadIdx.x >> 5;       // 0..7
  const int wm = wave & 3;                 // 0..3 (M)
  const int wn = wave >> 2;                // 0..1 (N)
  const int m0 = blockIdx.y * 128 + wm * 32;
  const int n0 = blockIdx.x * 64 + wn * 32;

  // A frag addressing: lane<16 -> K {0..7,16..23}; lane>=16 -> +8
  const unsigned short* aPtr0 = A + (size_t)(m0 + (lane & 15)) * Kd + ((lane >> 4) * 8);
  const unsigned short* aPtr1 = aPtr0 + (size_t)16 * Kd;
  // B frag addressing: lane holds col n; lane<16 -> K {0..7,16..23}; lane>=16 -> +8
  const unsigned short* bPtr0 = W + (size_t)(n0 + (lane & 15)) * Kd + ((lane >> 4) * 8);
  const unsigned short* bPtr1 = bPtr0 + (size_t)16 * Kd;

  v8f c00 = {}, c01 = {}, c10 = {}, c11 = {};
  for (int k0 = 0; k0 < Kd; k0 += 32) {
    // speculative prefetch one K-block ahead (global_prefetch_b8)
    __builtin_prefetch(aPtr0 + k0 + 32, 0, 3);
    __builtin_prefetch(aPtr1 + k0 + 32, 0, 3);
    __builtin_prefetch(bPtr0 + k0 + 32, 0, 3);
    __builtin_prefetch(bPtr1 + k0 + 32, 0, 3);
    Frag16 a0, a1, b0, b1;
    a0.q[0] = *(const uint4*)(aPtr0 + k0);
    a0.q[1] = *(const uint4*)(aPtr0 + k0 + 16);
    a1.q[0] = *(const uint4*)(aPtr1 + k0);
    a1.q[1] = *(const uint4*)(aPtr1 + k0 + 16);
    b0.q[0] = *(const uint4*)(bPtr0 + k0);
    b0.q[1] = *(const uint4*)(bPtr0 + k0 + 16);
    b1.q[0] = *(const uint4*)(bPtr1 + k0);
    b1.q[1] = *(const uint4*)(bPtr1 + k0 + 16);
    c00 = __builtin_amdgcn_wmma_f32_16x16x32_bf16(false, a0.v, false, b0.v, (short)0, c00, false, false);
    c01 = __builtin_amdgcn_wmma_f32_16x16x32_bf16(false, a0.v, false, b1.v, (short)0, c01, false, false);
    c10 = __builtin_amdgcn_wmma_f32_16x16x32_bf16(false, a1.v, false, b0.v, (short)0, c10, false, false);
    c11 = __builtin_amdgcn_wmma_f32_16x16x32_bf16(false, a1.v, false, b1.v, (short)0, c11, false, false);
  }
  const int outN  = n0 + (lane & 15);
  const int mBase = m0 + (lane >> 4) * 8;
#pragma unroll
  for (int v = 0; v < 8; ++v) {
    Y[(size_t)(mBase + v) * N + outN]           = c00[v];
    Y[(size_t)(mBase + v) * N + outN + 16]      = c01[v];
    Y[(size_t)(mBase + 16 + v) * N + outN]      = c10[v];
    Y[(size_t)(mBase + 16 + v) * N + outN + 16] = c11[v];
  }
}

// ---------------- WKV sequential scan, one block per (b,h) -----------------
__global__ void wkv_scan_kernel(const float* __restrict__ r, const float* __restrict__ k,
                                const float* __restrict__ v, const float* __restrict__ ew,
                                const float* __restrict__ u, const float* __restrict__ s0,
                                float* __restrict__ o, float* __restrict__ sOut) {
  int bh = blockIdx.x;                     // b*H + h
  int bb = bh / H_, h = bh % H_;
  int j = threadIdx.x;                     // 0..63, owns state column v=j
  __shared__ float rs[K_], ks[K_], es[K_], us[K_];
  us[j] = u[h * K_ + j];
  float S[K_];
  const float* sp = s0 + ((size_t)bb * H_ + h) * K_ * K_;
#pragma unroll
  for (int kk = 0; kk < K_; ++kk) S[kk] = sp[kk * K_ + j];
  __syncthreads();
  for (int t = 0; t < T_; ++t) {
    size_t base = ((size_t)(bb * T_ + t)) * C_ + h * K_;
    rs[j] = r[base + j]; ks[j] = k[base + j]; es[j] = ew[base + j];
    float vj = v[base + j];
    __syncthreads();
    float acc = 0.f;
#pragma unroll 8
    for (int kk = 0; kk < K_; ++kk) {
      float kv = ks[kk] * vj;
      acc += rs[kk] * (S[kk] + us[kk] * kv);
      S[kk] = es[kk] * S[kk] + kv;
    }
    o[base + j] = acc;
    __syncthreads();
  }
  float* so = sOut + ((size_t)bb * H_ + h) * K_ * K_;
#pragma unroll
  for (int kk = 0; kk < K_; ++kk) so[kk * K_ + j] = S[kk];
}

// ---------------- groupnorm(K=64) + lnx affine + *silu(g) -> bf16 z ---------
__global__ void gn_z_kernel(const float* __restrict__ o, const float* __restrict__ g,
                            const float* __restrict__ lnx_w, const float* __restrict__ lnx_b,
                            unsigned short* __restrict__ z) {
  int idx = blockIdx.x;                    // (b*T+t)*H + h
  int h = idx % H_, row = idx / H_;
  size_t base = (size_t)row * C_ + h * K_;
  int j = threadIdx.x;                     // 64 threads
  float val = o[base + j];
  __shared__ float red[K_];
  red[j] = val; __syncthreads();
  for (int off = 32; off > 0; off >>= 1) { if (j < off) red[j] += red[j + off]; __syncthreads(); }
  float mu = red[0] / K_; __syncthreads();
  red[j] = (val - mu) * (val - mu); __syncthreads();
  for (int off = 32; off > 0; off >>= 1) { if (j < off) red[j] += red[j + off]; __syncthreads(); }
  float inv = rsqrtf(red[0] / K_ + GN_EPS_);
  int c = h * K_ + j;
  float y  = (val - mu) * inv * lnx_w[c] + lnx_b[c];
  float gr = g[base + j];
  float sg = gr / (1.f + expf(-gr));       // silu
  z[base + j] = f2bf_bits(y * sg);
}

// ---------------- elementwise kernels --------------------------------------
__global__ void add_kernel(const float* __restrict__ a, const float* __restrict__ b,
                           float* __restrict__ y, size_t n) {
  size_t i = (size_t)blockIdx.x * blockDim.x + threadIdx.x;
  size_t st = (size_t)gridDim.x * blockDim.x;
  for (; i < n; i += st) y[i] = a[i] + b[i];
}

__global__ void cmix_kernel(const float* __restrict__ xn2, const float* __restrict__ cmst,
                            const float* __restrict__ fk, const float* __restrict__ fr,
                            unsigned short* __restrict__ xk2, unsigned short* __restrict__ xr2) {
  int row = blockIdx.x, t = row % T_, bb = row / T_;
  const float* xrow = xn2 + (size_t)row * C_;
  const float* pv = (t == 0) ? (cmst + (size_t)bb * C_) : (xrow - C_);
  for (int c = threadIdx.x; c < C_; c += blockDim.x) {
    float a = xrow[c], p = pv[c];
    float kf = fk[c], rf = fr[c];
    size_t o = (size_t)row * C_ + c;
    xk2[o] = f2bf_bits(p * kf + a * (1.f - kf));
    xr2[o] = f2bf_bits(p * rf + a * (1.f - rf));
  }
}

__global__ void relu2_kernel(const float* __restrict__ src, unsigned short* __restrict__ dst,
                             size_t n) {
  size_t i = (size_t)blockIdx.x * blockDim.x + threadIdx.x;
  size_t st = (size_t)gridDim.x * blockDim.x;
  for (; i < n; i += st) { float v = fmaxf(src[i], 0.f); dst[i] = f2bf_bits(v * v); }
}

__global__ void final_kernel(const float* __restrict__ xmid, const float* __restrict__ rraw,
                             const float* __restrict__ kv2, float* __restrict__ out, size_t n) {
  size_t i = (size_t)blockIdx.x * blockDim.x + threadIdx.x;
  size_t st = (size_t)gridDim.x * blockDim.x;
  for (; i < n; i += st) {
    float s = 1.f / (1.f + expf(-rraw[i]));
    out[i] = xmid[i] + s * kv2[i];
  }
}

// ---------------- host-side orchestration ----------------------------------
extern "C" void kernel_launch(void* const* d_in, const int* in_sizes, int n_in,
                              void* d_out, int out_size, void* d_ws, size_t ws_size,
                              hipStream_t stream) {
  (void)in_sizes; (void)n_in; (void)out_size; (void)ws_size;
  const float* x      = (const float*)d_in[0];
  const float* tshift = (const float*)d_in[1];
  const float* cmst   = (const float*)d_in[2];
  const float* tmst   = (const float*)d_in[3];
  const float* ln1w   = (const float*)d_in[4];
  const float* ln1b   = (const float*)d_in[5];
  const float* ln2w   = (const float*)d_in[6];
  const float* ln2b   = (const float*)d_in[7];
  const float* maa_x  = (const float*)d_in[8];
  const float* maa_w  = (const float*)d_in[9];
  const float* maa_k  = (const float*)d_in[10];
  const float* maa_v  = (const float*)d_in[11];
  const float* maa_r  = (const float*)d_in[12];
  const float* maa_g  = (const float*)d_in[13];
  const float* maa_w1 = (const float*)d_in[14];
  const float* maa_w2 = (const float*)d_in[15];
  const float* td     = (const float*)d_in[16];
  const float* td_w1  = (const float*)d_in[17];
  const float* td_w2  = (const float*)d_in[18];
  const float* u      = (const float*)d_in[19];
  const float* Wr     = (const float*)d_in[20];
  const float* Wk     = (const float*)d_in[21];
  const float* Wv     = (const float*)d_in[22];
  const float* Wg     = (const float*)d_in[23];
  const float* Wo     = (const float*)d_in[24];
  const float* lnxw   = (const float*)d_in[25];
  const float* lnxb   = (const float*)d_in[26];
  const float* fmk    = (const float*)d_in[27];
  const float* fmr    = (const float*)d_in[28];
  const float* fWk    = (const float*)d_in[29];
  const float* fWr    = (const float*)d_in[30];
  const float* fWv    = (const float*)d_in[31];

  float* out_x     = (float*)d_out;
  float* out_shift = out_x + (size_t)NTOK * C_;
  float* out_cm    = out_shift + (size_t)B_ * C_;
  float* out_wkv   = out_cm + (size_t)B_ * C_;

  char* ws = (char*)d_ws;
  size_t off = 0;
  auto alloc = [&](size_t bytes) -> void* {
    void* p = ws + off;
    off = (off + bytes + 255) & ~(size_t)255;
    return p;
  };
  const size_t NC = (size_t)NTOK * C_;
  const size_t NF = (size_t)NTOK * FF_;

  float* xn    = (float*)alloc(NC * 4);
  float* hh    = (float*)alloc((size_t)NTOK * 5 * DMIX * 4);
  float* xw    = (float*)alloc(NC * 4);
  float* tdh   = (float*)alloc((size_t)NTOK * DDEC * 4);
  float* ew    = (float*)alloc(NC * 4);
  unsigned short* xk  = (unsigned short*)alloc(NC * 2);
  unsigned short* xv  = (unsigned short*)alloc(NC * 2);
  unsigned short* xr  = (unsigned short*)alloc(NC * 2);
  unsigned short* xg  = (unsigned short*)alloc(NC * 2);
  unsigned short* bWr = (unsigned short*)alloc((size_t)C_ * C_ * 2);
  unsigned short* bWk = (unsigned short*)alloc((size_t)C_ * C_ * 2);
  unsigned short* bWv = (unsigned short*)alloc((size_t)C_ * C_ * 2);
  unsigned short* bWg = (unsigned short*)alloc((size_t)C_ * C_ * 2);
  unsigned short* bWo = (unsigned short*)alloc((size_t)C_ * C_ * 2);
  unsigned short* bfWk = (unsigned short*)alloc((size_t)FF_ * C_ * 2);
  unsigned short* bfWr = (unsigned short*)alloc((size_t)C_ * C_ * 2);
  unsigned short* bfWv = (unsigned short*)alloc((size_t)C_ * FF_ * 2);
  float* rbuf  = (float*)alloc(NC * 4);           // rbuf..gbuf later reused as kkraw
  float* kbuf  = (float*)alloc(NC * 4);
  float* vbuf  = (float*)alloc(NC * 4);
  float* gbuf  = (float*)alloc(NC * 4);
  float* obuf  = (float*)alloc(NC * 4);
  unsigned short* zbuf = (unsigned short*)alloc(NC * 2);
  float* owo   = (float*)alloc(NC * 4);
  float* xmid  = (float*)alloc(NC * 4);
  float* xn2   = (float*)alloc(NC * 4);
  unsigned short* xk2 = (unsigned short*)alloc(NC * 2);
  unsigned short* xr2 = (unsigned short*)alloc(NC * 2);
  unsigned short* kkbf = (unsigned short*)alloc(NF * 2);
  float* kv2   = (float*)alloc(NC * 4);
  float* rraw  = (float*)alloc(NC * 4);
  float* kkraw = rbuf;   // alias: 4*NC*4 bytes (67MB) >= NF*4 (58.7MB); r/k/v/g dead by then

  // --- weight conversion to bf16 (per call; deterministic) ---
  f2bf_kernel<<<2048, 256, 0, stream>>>(Wr,  bWr,  (size_t)C_ * C_);
  f2bf_kernel<<<2048, 256, 0, stream>>>(Wk,  bWk,  (size_t)C_ * C_);
  f2bf_kernel<<<2048, 256, 0, stream>>>(Wv,  bWv,  (size_t)C_ * C_);
  f2bf_kernel<<<2048, 256, 0, stream>>>(Wg,  bWg,  (size_t)C_ * C_);
  f2bf_kernel<<<2048, 256, 0, stream>>>(Wo,  bWo,  (size_t)C_ * C_);
  f2bf_kernel<<<2048, 256, 0, stream>>>(fWk, bfWk, (size_t)FF_ * C_);
  f2bf_kernel<<<2048, 256, 0, stream>>>(fWr, bfWr, (size_t)C_ * C_);
  f2bf_kernel<<<2048, 256, 0, stream>>>(fWv, bfWv, (size_t)C_ * FF_);

  // --- time-mix front end ---
  ln_kernel<<<NTOK, 256, 0, stream>>>(x, ln1w, ln1b, xn, out_shift);
  mix_hh_kernel<<<NTOK, 160, 0, stream>>>(xn, tshift, maa_x, maa_w1, hh);
  mix_apply_kernel<<<NTOK, 256, 0, stream>>>(xn, tshift, hh, maa_w2, maa_w, maa_k,
                                             maa_v, maa_r, maa_g, xw, xk, xv, xr, xg);
  td1_kernel<<<NTOK, 64, 0, stream>>>(xw, td_w1, tdh);
  wdecay_kernel<<<NTOK, 256, 0, stream>>>(tdh, td_w2, td, ew);

  // --- big GEMMs (bf16 WMMA, fp32 accumulate); block tile 128x64 -----------
  dim3 gCC(C_ / 64, NTOK / 128);
  gemm_bf16_wmma<<<gCC, 256, 0, stream>>>(xr, bWr, rbuf, NTOK, C_, C_);
  gemm_bf16_wmma<<<gCC, 256, 0, stream>>>(xk, bWk, kbuf, NTOK, C_, C_);
  gemm_bf16_wmma<<<gCC, 256, 0, stream>>>(xv, bWv, vbuf, NTOK, C_, C_);
  gemm_bf16_wmma<<<gCC, 256, 0, stream>>>(xg, bWg, gbuf, NTOK, C_, C_);

  // --- sequential WKV scan + groupnorm + gating ---
  wkv_scan_kernel<<<B_ * H_, K_, 0, stream>>>(rbuf, kbuf, vbuf, ew, u, tmst, obuf, out_wkv);
  gn_z_kernel<<<NTOK * H_, K_, 0, stream>>>(obuf, gbuf, lnxw, lnxb, zbuf);

  // --- output projection + residual ---
  gemm_bf16_wmma<<<gCC, 256, 0, stream>>>(zbuf, bWo, owo, NTOK, C_, C_);
  add_kernel<<<1024, 256, 0, stream>>>(x, owo, xmid, NC);

  // --- channel mix ---
  ln_kernel<<<NTOK, 256, 0, stream>>>(xmid, ln2w, ln2b, xn2, out_cm);
  cmix_kernel<<<NTOK, 256, 0, stream>>>(xn2, cmst, fmk, fmr, xk2, xr2);

  dim3 gFC(FF_ / 64, NTOK / 128);
  gemm_bf16_wmma<<<gFC, 256, 0, stream>>>(xk2, bfWk, kkraw, NTOK, FF_, C_);
  relu2_kernel<<<2048, 256, 0, stream>>>(kkraw, kkbf, NF);
  gemm_bf16_wmma<<<gCC, 256, 0, stream>>>(kkbf, bfWv, kv2, NTOK, C_, FF_);
  gemm_bf16_wmma<<<gCC, 256, 0, stream>>>(xr2, bfWr, rraw, NTOK, C_, C_);

  final_kernel<<<1024, 256, 0, stream>>>(xmid, rraw, kv2, out_x, NC);
}
